// CrossMHA_13666585936399
// MI455X (gfx1250) — matrix operand order repaired and checked by reading, not compiled, not run on hardware
//
#include <hip/hip_runtime.h>

typedef __attribute__((ext_vector_type(16))) _Float16 v16h;
typedef __attribute__((ext_vector_type(8)))  _Float16 v8h;
typedef __attribute__((ext_vector_type(8)))  float    v8f;
typedef __attribute__((ext_vector_type(4)))  unsigned v4u;

union V16 { v16h v; v8h h[2]; };

#define B_   2
#define S_   2048
#define DIM_ 1024
#define H_   16
#define DK_  64
#define SW_  (S_ / 32)          // mask words per query row
#define NEGV (-1.0e10f)

// ---------------------------------------------------------------------------
// fp32 -> f16 conversion (inputs are fp32; WMMA wants f16)
// ---------------------------------------------------------------------------
__global__ void cvt_f32_f16_kernel(const float* __restrict__ in,
                                   _Float16* __restrict__ out, int n) {
  int i = blockIdx.x * blockDim.x + threadIdx.x;
  if (i < n) out[i] = (_Float16)in[i];
}

// ---------------------------------------------------------------------------
// Pack int32 0/1 mask -> 1 bit per key, stored TRANSPOSED as [b][kw][q] so
// the 8 words a lane needs per key-chunk are contiguous (2x b128 loads).
// ---------------------------------------------------------------------------
__global__ void pack_mask_kernel(const int* __restrict__ mask,
                                 unsigned* __restrict__ bitsT) {
  int i = blockIdx.x * blockDim.x + threadIdx.x;   // flat (b, q, kw)
  const int* src = mask + (size_t)i * 32;
  unsigned w = 0;
#pragma unroll
  for (int j = 0; j < 32; ++j) w |= (src[j] != 0 ? 1u : 0u) << j;
  int kw = i & (SW_ - 1);
  int q  = (i >> 6) & (S_ - 1);
  int b  = i >> 17;                                // 6 + 11
  bitsT[(size_t)(b * SW_ + kw) * S_ + q] = w;
}

// ---------------------------------------------------------------------------
// GEMM: C[M,N] = A[M,1024] @ W[N,1024]^T   (f16 in, f32 accumulate via WMMA)
// Each wave computes a 16x64 tile; all 4 B-tiles loaded before the WMMA group.
// MODE 0: store f16 * 0.125 (fold 1/sqrt(DK)), head-permute -> Qh[b,h,s,d]
// MODE 1: split KV: n<1024 -> Kh[b,h,s,d]; n>=1024 -> Vt[b,h,d,s] (transposed)
// MODE 2: store f32 to outf[m*N+n]  (final @ Wo^T)
// ---------------------------------------------------------------------------
template<int MODE>
__global__ void gemm_wmma_kernel(const _Float16* __restrict__ A,
                                 const _Float16* __restrict__ W,
                                 _Float16* __restrict__ out0,
                                 _Float16* __restrict__ out1,
                                 float* __restrict__ outf,
                                 int N) {
  const int K = DIM_;
  int wave = threadIdx.x >> 5;
  int lane = threadIdx.x & 31;
  int hlf  = lane >> 4;      // which 16-lane half
  int l16  = lane & 15;
  int mrow  = blockIdx.y * 16;
  int nbase = blockIdx.x * 256 + wave * 64;

  const _Float16* Arow = A + (size_t)(mrow + l16) * K;
  v8f acc[4] = {};

  for (int k = 0; k < K; k += 32) {
    // A-tile: lane l16 holds row l16; elems 0..15 = K(16h..16h+15)
    V16 a;
    a.h[0] = *(const v8h*)(Arow + k + 16 * hlf);
    a.h[1] = *(const v8h*)(Arow + k + 16 * hlf + 8);
    // batch-load all 4 B-tiles (lane = column n of B = row n of W)
    V16 bm[4];
#pragma unroll
    for (int t = 0; t < 4; ++t) {
      const _Float16* wr = W + (size_t)(nbase + t * 16 + l16) * K + k + 16 * hlf;
      bm[t].h[0] = *(const v8h*)(wr);
      bm[t].h[1] = *(const v8h*)(wr + 8);
    }
#pragma unroll
    for (int t = 0; t < 4; ++t)
      acc[t] = __builtin_amdgcn_wmma_f32_16x16x32_f16(
          false, a.v, false, bm[t].v, (short)0, acc[t], false, false);
  }

  // Epilogue: C layout = lane(l16)=col, VGPR r = row r + 8*hlf
#pragma unroll
  for (int t = 0; t < 4; ++t) {
#pragma unroll
    for (int r = 0; r < 8; ++r) {
      int m = mrow + r + 8 * hlf;
      int n = nbase + t * 16 + l16;
      float v = acc[t][r];
      if (MODE == 2) {
        outf[(size_t)m * N + n] = v;
      } else {
        int b = m >> 11;          // m / S_
        int s = m & (S_ - 1);
        if (MODE == 0) {
          int hh = n & 15, d = n >> 4;   // dim index = d*H + h
          out0[((size_t)(b * H_ + hh) * S_ + s) * DK_ + d] =
              (_Float16)(v * 0.125f);    // fold 1/sqrt(DK) into Q
        } else {
          _Float16 hv = (_Float16)v;
          if (n < DIM_) {
            int hh = n & 15, d = n >> 4;
            out0[((size_t)(b * H_ + hh) * S_ + s) * DK_ + d] = hv;
          } else {
            int nn = n - DIM_;
            int hh = nn & 15, d = nn >> 4;
            out1[((size_t)(b * H_ + hh) * DK_ + d) * S_ + s] = hv; // V^T
          }
        }
      }
    }
  }
}

// ---------------------------------------------------------------------------
// Flash attention per (b, h): each wave owns one 16-query tile, streams keys
// in chunks of 32. Scores stay in VGPRs; online softmax over the key axis.
// ---------------------------------------------------------------------------
__global__ void attn_flash_kernel(const _Float16* __restrict__ Qh,
                                  const _Float16* __restrict__ Kh,
                                  const _Float16* __restrict__ Vt,
                                  const unsigned* __restrict__ bitsT,
                                  _Float16* __restrict__ Ob) {
  __shared__ _Float16 Plds[4 * 16 * 32];   // per-wave 16x32 P staging
  int wave = threadIdx.x >> 5;
  int lane = threadIdx.x & 31;
  int hlf  = lane >> 4;
  int l16  = lane & 15;
  int b = blockIdx.z;
  int h = blockIdx.y;
  int qbase = (blockIdx.x * 4 + wave) * 16;

  const _Float16* Qp = Qh + ((size_t)(b * H_ + h) * S_ + qbase) * DK_;
  const _Float16* Kp = Kh + (size_t)(b * H_ + h) * S_ * DK_;
  const _Float16* Vp = Vt + (size_t)(b * H_ + h) * DK_ * S_;
  const unsigned* mbT = bitsT + (size_t)b * SW_ * S_ + qbase + 8 * hlf;
  _Float16* pl = Plds + wave * (16 * 32);

  unsigned bit0 = 1u << l16;         // tile-0 key bit for this lane
  unsigned bit1 = 1u << (16 + l16);  // tile-1 key bit

  // Q as two A-tiles (d-chunks of 32); Q already carries the 1/8 scale
  V16 qa[2];
  {
    const _Float16* qr = Qp + (size_t)l16 * DK_;
#pragma unroll
    for (int c = 0; c < 2; ++c) {
      qa[c].h[0] = *(const v8h*)(qr + 32 * c + 16 * hlf);
      qa[c].h[1] = *(const v8h*)(qr + 32 * c + 16 * hlf + 8);
    }
  }

  float mi[8], li[8];
  v8f oacc[4] = {};
#pragma unroll
  for (int r = 0; r < 8; ++r) { mi[r] = -1.0e30f; li[r] = 0.0f; }

  for (int kb = 0; kb < S_; kb += 32) {
    // mask bit-words for this chunk: 8 contiguous words -> two b128 loads
    const unsigned* mwp = mbT + (size_t)(kb >> 5) * S_;
    v4u w0 = *(const v4u*)(mwp);
    v4u w1 = *(const v4u*)(mwp + 4);

    // prefetch next chunk of K rows and V^T rows (global_prefetch_b8)
    if (kb + 32 < S_) {
      __builtin_prefetch(Kp + (size_t)(kb + 32 + lane) * DK_, 0, 3);
      __builtin_prefetch(Vp + (size_t)lane * S_ + kb + 32, 0, 3);
      __builtin_prefetch(Vp + (size_t)(32 + lane) * S_ + kb + 32, 0, 3);
    }

    // ---- batch-load all K B-tiles, then run the 4 score WMMAs
    V16 kbm[2][2];
#pragma unroll
    for (int t = 0; t < 2; ++t) {
#pragma unroll
      for (int c = 0; c < 2; ++c) {
        const _Float16* kr =
            Kp + (size_t)(kb + t * 16 + l16) * DK_ + 32 * c + 16 * hlf;
        kbm[t][c].h[0] = *(const v8h*)(kr);
        kbm[t][c].h[1] = *(const v8h*)(kr + 8);
      }
    }
    v8f sc[2] = {};
#pragma unroll
    for (int t = 0; t < 2; ++t)
#pragma unroll
      for (int c = 0; c < 2; ++c)
        sc[t] = __builtin_amdgcn_wmma_f32_16x16x32_f16(
            false, qa[c].v, false, kbm[t][c].v, (short)0, sc[t], false, false);

    // ---- mask (scale already folded into Q)
    float p0[8], p1[8];
#pragma unroll
    for (int r = 0; r < 8; ++r) {
      unsigned mw = (r < 4) ? w0[r] : w1[r - 4];
      float v0 = sc[0][r];
      float v1 = sc[1][r];
      if (!(mw & bit0)) v0 = NEGV;
      if (!(mw & bit1)) v1 = NEGV;
      p0[r] = v0; p1[r] = v1;
    }
    // ---- online softmax (row = fixed VGPR idx across a 16-lane group)
#pragma unroll
    for (int r = 0; r < 8; ++r) {
      float pm = fmaxf(p0[r], p1[r]);
      pm = fmaxf(pm, __shfl_xor(pm, 1, 32));
      pm = fmaxf(pm, __shfl_xor(pm, 2, 32));
      pm = fmaxf(pm, __shfl_xor(pm, 4, 32));
      pm = fmaxf(pm, __shfl_xor(pm, 8, 32));
      float mnew  = fmaxf(mi[r], pm);
      float alpha = __expf(mi[r] - mnew);
      float e0 = __expf(p0[r] - mnew);
      float e1 = __expf(p1[r] - mnew);
      p0[r] = e0; p1[r] = e1;
      float ps = e0 + e1;
      ps += __shfl_xor(ps, 1, 32);
      ps += __shfl_xor(ps, 2, 32);
      ps += __shfl_xor(ps, 4, 32);
      ps += __shfl_xor(ps, 8, 32);
      li[r] = li[r] * alpha + ps;
      mi[r] = mnew;
#pragma unroll
      for (int t = 0; t < 4; ++t) oacc[t][r] *= alpha;
    }
    // ---- P: C-layout -> LDS (row-major 16x32) -> A-layout
    __syncthreads();
#pragma unroll
    for (int r = 0; r < 8; ++r) {
      int q = r + 8 * hlf;
      pl[q * 32 + l16]      = (_Float16)p0[r];
      pl[q * 32 + 16 + l16] = (_Float16)p1[r];
    }
    __syncthreads();
    V16 pa;
    pa.h[0] = *(const v8h*)(pl + l16 * 32 + 16 * hlf);
    pa.h[1] = *(const v8h*)(pl + l16 * 32 + 16 * hlf + 8);
    // ---- batch-load all V B-tiles, then run the 4 PV WMMAs
    V16 vb[4];
#pragma unroll
    for (int t = 0; t < 4; ++t) {
      const _Float16* vr = Vp + (size_t)(t * 16 + l16) * S_ + kb + 16 * hlf;
      vb[t].h[0] = *(const v8h*)(vr);
      vb[t].h[1] = *(const v8h*)(vr + 8);
    }
#pragma unroll
    for (int t = 0; t < 4; ++t)
      oacc[t] = __builtin_amdgcn_wmma_f32_16x16x32_f16(
          false, pa.v, false, vb[t].v, (short)0, oacc[t], false, false);
  }

  // ---- finalize: multiply by reciprocal row sums, scatter to [b,s,d*H+h]
#pragma unroll
  for (int r = 0; r < 8; ++r) {
    float rl = 1.0f / li[r];
    int q = qbase + r + 8 * hlf;
#pragma unroll
    for (int t = 0; t < 4; ++t) {
      int d = t * 16 + l16;
      Ob[((size_t)b * S_ + q) * DIM_ + d * 16 + h] =
          (_Float16)(oacc[t][r] * rl);
    }
  }
}

// ---------------------------------------------------------------------------
extern "C" void kernel_launch(void* const* d_in, const int* in_sizes, int n_in,
                              void* d_out, int out_size, void* d_ws, size_t ws_size,
                              hipStream_t stream) {
  (void)in_sizes; (void)n_in; (void)out_size; (void)ws_size;
  const float* dec = (const float*)d_in[0];
  const float* enc = (const float*)d_in[1];
  const float* Wq  = (const float*)d_in[2];
  const float* Wkv = (const float*)d_in[3];
  const float* Wo  = (const float*)d_in[4];
  const int*  mask = (const int*)d_in[5];
  float* out = (float*)d_out;

  const size_t MEG = 1024ull * 1024ull;   // element blocks
  _Float16* w = (_Float16*)d_ws;
  _Float16* dec16 = w + 0 * MEG;          // 4M
  _Float16* enc16 = w + 4 * MEG;          // 4M
  _Float16* Wq16  = w + 8 * MEG;          // 1M
  _Float16* Wkv16 = w + 9 * MEG;          // 2M
  _Float16* Wo16  = w + 11 * MEG;         // 1M
  _Float16* Qh    = w + 12 * MEG;         // 4M  [b,h,s,d]  (pre-scaled by 1/8)
  _Float16* Kh    = w + 16 * MEG;         // 4M  [b,h,s,d]
  _Float16* Vt    = w + 20 * MEG;         // 4M  [b,h,d,s]
  _Float16* Ob    = w + 24 * MEG;         // 4M  [b,s,dim]
  unsigned* mbits = (unsigned*)(w + 28 * MEG);  // 256K words = 1MB (total 57MB)

  // fp32 -> f16
  cvt_f32_f16_kernel<<<dim3(4 * MEG / 256), dim3(256), 0, stream>>>(dec, dec16, (int)(4 * MEG));
  cvt_f32_f16_kernel<<<dim3(4 * MEG / 256), dim3(256), 0, stream>>>(enc, enc16, (int)(4 * MEG));
  cvt_f32_f16_kernel<<<dim3(1 * MEG / 256), dim3(256), 0, stream>>>(Wq,  Wq16,  (int)(1 * MEG));
  cvt_f32_f16_kernel<<<dim3(2 * MEG / 256), dim3(256), 0, stream>>>(Wkv, Wkv16, (int)(2 * MEG));
  cvt_f32_f16_kernel<<<dim3(1 * MEG / 256), dim3(256), 0, stream>>>(Wo,  Wo16,  (int)(1 * MEG));

  // mask -> transposed bit words [b][kw][q]
  pack_mask_kernel<<<dim3((B_ * S_ * SW_) / 256), dim3(256), 0, stream>>>(mask, mbits);

  // Q = dec @ Wq^T  (head-permuted, pre-scaled store)
  gemm_wmma_kernel<0><<<dim3(DIM_ / 256, (B_ * S_) / 16), dim3(128), 0, stream>>>(
      dec16, Wq16, Qh, nullptr, nullptr, DIM_);
  // KV = enc @ Wkv^T  (K head-permuted, V head-permuted + transposed)
  gemm_wmma_kernel<1><<<dim3(2 * DIM_ / 256, (B_ * S_) / 16), dim3(128), 0, stream>>>(
      enc16, Wkv16, Kh, Vt, nullptr, 2 * DIM_);
  // fused masked-softmax attention
  attn_flash_kernel<<<dim3(S_ / 64, H_, B_), dim3(128), 0, stream>>>(Qh, Kh, Vt, mbits, Ob);
  // out = O @ Wo^T  (fp32 result)
  gemm_wmma_kernel<2><<<dim3(DIM_ / 256, (B_ * S_) / 16), dim3(128), 0, stream>>>(
      Ob, Wo16, nullptr, nullptr, out, DIM_);
}
